// XiSSA_21148418966304
// MI455X (gfx1250) — 16-bit and 8-bit matrix operand order repaired and checked by reading, compiled, not run on hardware
//
#include <hip/hip_runtime.h>
#include <math.h>

typedef __attribute__((ext_vector_type(16))) _Float16 v16h;
typedef __attribute__((ext_vector_type(8)))  _Float16 v8h;
typedef __attribute__((ext_vector_type(8)))  float    v8f;
typedef __attribute__((ext_vector_type(8)))  int      v8i;
typedef __attribute__((ext_vector_type(4)))  int      v4i;
typedef __attribute__((ext_vector_type(2)))  int      v2i;

#define T_  4
#define B_  16
#define N_  256
#define C_  512
#define H_  8
#define HD_ 64
#define M_  (T_ * B_ * N_)
#define EPS_ 1e-5f

union I8x { v8i v; int i[8]; };

__device__ __forceinline__ v8i load_a8(const unsigned char* row, int g) {
  I8x r;
  const unsigned char* p = row + 16 * g;
  v2i t0 = *(const v2i*)(p);
  v2i t1 = *(const v2i*)(p + 8);
  v2i t2 = *(const v2i*)(p + 32);
  v2i t3 = *(const v2i*)(p + 40);
  r.i[0] = t0[0]; r.i[1] = t0[1];
  r.i[2] = t1[0]; r.i[3] = t1[1];
  r.i[4] = t2[0]; r.i[5] = t2[1];
  r.i[6] = t3[0]; r.i[7] = t3[1];
  return r.v;
}

__device__ __forceinline__ v8i load_b8(const unsigned char* row, int g) {
  I8x r;
  const unsigned char* p = row + 16 * g;
  v4i t0 = *(const v4i*)(p);
  v4i t1 = *(const v4i*)(p + 32);
  r.i[0] = t0[0]; r.i[1] = t0[1]; r.i[2] = t0[2]; r.i[3] = t0[3];
  r.i[4] = t1[0]; r.i[5] = t1[1]; r.i[6] = t1[2]; r.i[7] = t1[3];
  return r.v;
}

__global__ void split_hi_lo(const float* __restrict__ X,
                            _Float16* __restrict__ hi,
                            _Float16* __restrict__ lo, int n) {
  int i = blockIdx.x * blockDim.x + threadIdx.x;
  if (i >= n) return;
  float x = X[i];
  _Float16 h = (_Float16)x;
  hi[i] = h;
  lo[i] = (_Float16)(x - (float)h);
}

template <int NPROD>
__global__ void gemm16_f16split(const _Float16* __restrict__ Ahi,
                                const _Float16* __restrict__ Alo,
                                const _Float16* __restrict__ Whi,
                                const _Float16* __restrict__ Wlo,
                                float* __restrict__ Y) {
  const int lane = threadIdx.x & 31;
  const int wid  = threadIdx.x >> 5;
  const int g    = lane >> 4;
  const int mn   = lane & 15;
  const int strip = blockIdx.x * 8 + wid;
  const int tm    = strip >> 3;
  const int tn0   = (strip & 7) * 4;

  const _Float16* arow_hi = Ahi + (size_t)(tm * 16 + mn) * C_;
  const _Float16* arow_lo = (NPROD == 3) ? (Alo + (size_t)(tm * 16 + mn) * C_) : arow_hi;

  v8f acc[4];
#pragma unroll
  for (int j = 0; j < 4; ++j) acc[j] = (v8f){0.f, 0.f, 0.f, 0.f, 0.f, 0.f, 0.f, 0.f};

  for (int k = 0; k < C_; k += 32) {
    const int ka0 = k + 16 * g;
    const int ka1 = k + 16 * g + 8;
    v8h a0 = *(const v8h*)(arow_hi + ka0);
    v8h a1 = *(const v8h*)(arow_hi + ka1);
    v16h ah = __builtin_shufflevector(a0, a1, 0,1,2,3,4,5,6,7,8,9,10,11,12,13,14,15);
    v16h al;
    if (NPROD == 3) {
      v8h l0 = *(const v8h*)(arow_lo + ka0);
      v8h l1 = *(const v8h*)(arow_lo + ka1);
      al = __builtin_shufflevector(l0, l1, 0,1,2,3,4,5,6,7,8,9,10,11,12,13,14,15);
    }
#pragma unroll
    for (int j = 0; j < 4; ++j) {
      const size_t boff = (size_t)((tn0 + j) * 16 + mn) * C_ + k + 16 * g;
      v16h bh = *(const v16h*)(Whi + boff);
      v16h bl = *(const v16h*)(Wlo + boff);
      acc[j] = __builtin_amdgcn_wmma_f32_16x16x32_f16(false, ah, false, bh, (short)0, acc[j], false, false);
      acc[j] = __builtin_amdgcn_wmma_f32_16x16x32_f16(false, ah, false, bl, (short)0, acc[j], false, false);
      if (NPROD == 3)
        acc[j] = __builtin_amdgcn_wmma_f32_16x16x32_f16(false, al, false, bh, (short)0, acc[j], false, false);
    }
  }
#pragma unroll
  for (int j = 0; j < 4; ++j) {
#pragma unroll
    for (int r = 0; r < 8; ++r) {
      int m = tm * 16 + r + 8 * g;
      Y[(size_t)m * C_ + (tn0 + j) * 16 + mn] = acc[j][r];
    }
  }
}

__global__ void col_stats(const float* __restrict__ Y,
                          float* __restrict__ psum, float* __restrict__ psq) {
  __shared__ float ls[256];
  __shared__ float ls2[256];
  const int c  = blockIdx.x * 64 + (threadIdx.x & 63);
  const int r0 = blockIdx.y * 256 + (threadIdx.x >> 6);
  float s = 0.f, s2 = 0.f;
  for (int i = 0; i < 64; ++i) {
    float y = Y[(size_t)(r0 + i * 4) * C_ + c];
    s += y; s2 += y * y;
  }
  ls[threadIdx.x] = s; ls2[threadIdx.x] = s2;
  __syncthreads();
  if (threadIdx.x < 64) {
    int t = threadIdx.x;
    float a  = ((ls[t]  + ls[t + 64])  + ls[t + 128])  + ls[t + 192];
    float b  = ((ls2[t] + ls2[t + 64]) + ls2[t + 128]) + ls2[t + 192];
    psum[(size_t)blockIdx.y * C_ + c] = a;
    psq [(size_t)blockIdx.y * C_ + c] = b;
  }
}

__global__ void bn_finalize(const float* __restrict__ psum, const float* __restrict__ psq,
                            const float* __restrict__ gamma, const float* __restrict__ beta,
                            float* __restrict__ scale, float* __restrict__ shift) {
  int c = blockIdx.x * 256 + threadIdx.x;
  float s = 0.f, s2 = 0.f;
  for (int j = 0; j < 64; ++j) { s += psum[(size_t)j * C_ + c]; s2 += psq[(size_t)j * C_ + c]; }
  const float invM = 1.f / (float)M_;
  float m   = s * invM;
  float var = s2 * invM - m * m;
  float inv = 1.f / sqrtf(var + EPS_);
  float sc  = gamma[c] * inv;
  scale[c] = sc;
  shift[c] = beta[c] - m * sc;
}

__global__ void bn_lif_qkv(const float* __restrict__ Y,
                           const float* __restrict__ scale,
                           const float* __restrict__ shift,
                           unsigned char* __restrict__ S, int vtrans) {
#pragma clang fp contract(off)
  int idx = blockIdx.x * blockDim.x + threadIdx.x;
  int c = idx & (C_ - 1);
  int n = (idx >> 9) & (N_ - 1);
  int b = idx >> 17;
  int hh = c >> 6, d = c & 63;
  float sc = scale[c], sh = shift[c];
  float v = 0.f;
  for (int t = 0; t < T_; ++t) {
    float x = Y[(size_t)((t * B_ + b) * N_ + n) * C_ + c] * sc + sh;
    v = v + (x - v) * 0.5f;
    float s = (v >= 1.0f) ? 1.f : 0.f;
    size_t o = vtrans
      ? (((size_t)(t * B_ + b) * H_ + hh) * HD_ + d) * N_ + n
      : (((size_t)(t * B_ + b) * H_ + hh) * N_  + n) * HD_ + d;
    S[o] = (unsigned char)s;
    v = v * (1.f - s);
  }
}

__global__ void spike_attention(const unsigned char* __restrict__ Q,
                                const unsigned char* __restrict__ Kk,
                                const unsigned char* __restrict__ Vt,
                                float* __restrict__ O) {
  __shared__ unsigned char attn[N_ * N_];
  const int lane = threadIdx.x & 31;
  const int wid  = threadIdx.x >> 5;
  const int g    = lane >> 4;
  const int mn   = lane & 15;
  const int tbh  = blockIdx.x;
  const int h    = tbh & 7;
  const int tb   = tbh >> 3;
  const unsigned char* qb = Q  + (size_t)tbh * N_ * HD_;
  const unsigned char* kb = Kk + (size_t)tbh * N_ * HD_;
  const unsigned char* vb = Vt + (size_t)tbh * HD_ * N_;

  for (int i = 0; i < 32; ++i) {
    int tile = wid * 32 + i;
    int tm = tile >> 4, tn = tile & 15;
    v8i a = load_a8(qb + (size_t)(tm * 16 + mn) * HD_, g);
    v8i b = load_b8(kb + (size_t)(tn * 16 + mn) * HD_, g);
    v8i acc = {0, 0, 0, 0, 0, 0, 0, 0};
    acc = __builtin_amdgcn_wmma_i32_16x16x64_iu8(false, a, false, b, acc, false, false);
#pragma unroll
    for (int r = 0; r < 8; ++r)
      attn[(tm * 16 + r + 8 * g) * N_ + tn * 16 + mn] = (unsigned char)acc[r];
  }
  __syncthreads();

  for (int i = 0; i < 8; ++i) {
    int tile = wid * 8 + i;
    int tm = tile >> 2, tn = tile & 3;
    v8i acc = {0, 0, 0, 0, 0, 0, 0, 0};
#pragma unroll
    for (int kk = 0; kk < 4; ++kk) {
      v8i a = load_a8(&attn[(tm * 16 + mn) * N_ + kk * 64], g);
      v8i b = load_b8(vb + (size_t)(tn * 16 + mn) * N_ + kk * 64, g);
      acc = __builtin_amdgcn_wmma_i32_16x16x64_iu8(false, a, false, b, acc, false, false);
    }
#pragma unroll
    for (int r = 0; r < 8; ++r) {
      int n = tm * 16 + r + 8 * g;
      int d = tn * 16 + mn;
      O[((size_t)tb * N_ + n) * C_ + h * HD_ + d] = 0.125f * (float)acc[r];
    }
  }
}

__global__ void lif_o(const float* __restrict__ O, _Float16* __restrict__ S) {
#pragma clang fp contract(off)
  int idx = blockIdx.x * blockDim.x + threadIdx.x;
  int c = idx & (C_ - 1);
  int n = (idx >> 9) & (N_ - 1);
  int b = idx >> 17;
  float v = 0.f;
  for (int t = 0; t < T_; ++t) {
    size_t o = (size_t)((t * B_ + b) * N_ + n) * C_ + c;
    float x = O[o];
    v = v + (x - v) * 0.5f;
    float s = (v >= 0.5f) ? 1.f : 0.f;
    S[o] = (_Float16)s;
    v = v * (1.f - s);
  }
}

__global__ void bn_lif_out(const float* __restrict__ Y,
                           const float* __restrict__ scale,
                           const float* __restrict__ shift,
                           float* __restrict__ out) {
#pragma clang fp contract(off)
  int idx = blockIdx.x * blockDim.x + threadIdx.x;
  int c = idx & (C_ - 1);
  int n = (idx >> 9) & (N_ - 1);
  int b = idx >> 17;
  float sc = scale[c], sh = shift[c];
  float v = 0.f;
  for (int t = 0; t < T_; ++t) {
    size_t o = (size_t)((t * B_ + b) * N_ + n) * C_ + c;
    float x = Y[o] * sc + sh;
    v = v + (x - v) * 0.5f;
    float s = (v >= 1.0f) ? 1.f : 0.f;
    out[o] = s;
    v = v * (1.f - s);
  }
}

static constexpr size_t OFF_XHI  = 0;
static constexpr size_t OFF_XLO  = OFF_XHI + (size_t)M_ * C_ * 2;
static constexpr size_t OFF_WHI  = OFF_XLO + (size_t)M_ * C_ * 2;
static constexpr size_t OFF_WLO  = OFF_WHI + 4ull * C_ * C_ * 2;
static constexpr size_t OFF_YQ   = OFF_WLO + 4ull * C_ * C_ * 2;
static constexpr size_t OFF_YK   = OFF_YQ + (size_t)M_ * C_ * 4;
static constexpr size_t OFF_YV   = OFF_YK + (size_t)M_ * C_ * 4;
static constexpr size_t OFF_SQ   = OFF_YV + (size_t)M_ * C_ * 4;
static constexpr size_t OFF_SK   = OFF_SQ + (size_t)M_ * C_;
static constexpr size_t OFF_SVT  = OFF_SK + (size_t)M_ * C_;
static constexpr size_t OFF_PSUM = OFF_SVT + (size_t)M_ * C_;
static constexpr size_t OFF_PSQ  = OFF_PSUM + 64ull * C_ * 4;
static constexpr size_t OFF_SS   = OFF_PSQ + 64ull * C_ * 4;

extern "C" void kernel_launch(void* const* d_in, const int* in_sizes, int n_in,
                              void* d_out, int out_size, void* d_ws, size_t ws_size,
                              hipStream_t stream) {
  (void)in_sizes; (void)n_in; (void)out_size; (void)ws_size;
  const float* x = (const float*)d_in[0];
  const float* w[4]     = {(const float*)d_in[1], (const float*)d_in[2],
                           (const float*)d_in[3], (const float*)d_in[4]};
  const float* gamma[4] = {(const float*)d_in[5], (const float*)d_in[7],
                           (const float*)d_in[9], (const float*)d_in[11]};
  const float* beta[4]  = {(const float*)d_in[6], (const float*)d_in[8],
                           (const float*)d_in[10], (const float*)d_in[12]};

  char* ws = (char*)d_ws;
  _Float16* xhi = (_Float16*)(ws + OFF_XHI);
  _Float16* xlo = (_Float16*)(ws + OFF_XLO);
  _Float16* whi = (_Float16*)(ws + OFF_WHI);
  _Float16* wlo = (_Float16*)(ws + OFF_WLO);
  float* Y[3]   = {(float*)(ws + OFF_YQ), (float*)(ws + OFF_YK), (float*)(ws + OFF_YV)};
  unsigned char* Sq  = (unsigned char*)(ws + OFF_SQ);
  unsigned char* Sk  = (unsigned char*)(ws + OFF_SK);
  unsigned char* SvT = (unsigned char*)(ws + OFF_SVT);
  float* psum = (float*)(ws + OFF_PSUM);
  float* psq  = (float*)(ws + OFF_PSQ);
  float* ssbase = (float*)(ws + OFF_SS);
  float* O      = (float*)(ws + OFF_YQ);
  _Float16* Sp  = (_Float16*)(ws + OFF_YK);
  float* Yp     = (float*)(ws + OFF_YV);

  const dim3 blk(256);
  const int nX = M_ * C_;
  const int nW = C_ * C_;
  const int gemmGrid = (M_ / 16) * (C_ / 64) / 8;
  const int elemGrid = (B_ * N_ * C_) / 256;

  split_hi_lo<<<nX / 256, blk, 0, stream>>>(x, xhi, xlo, nX);
  for (int i = 0; i < 4; ++i)
    split_hi_lo<<<nW / 256, blk, 0, stream>>>(w[i], whi + (size_t)i * nW, wlo + (size_t)i * nW, nW);

  unsigned char* Sdst[3] = {Sq, Sk, SvT};
  for (int i = 0; i < 3; ++i) {
    gemm16_f16split<3><<<gemmGrid, blk, 0, stream>>>(xhi, xlo,
        whi + (size_t)i * nW, wlo + (size_t)i * nW, Y[i]);
    col_stats<<<dim3(8, 64), blk, 0, stream>>>(Y[i], psum, psq);
    float* sc = ssbase + i * 2 * C_;
    bn_finalize<<<2, blk, 0, stream>>>(psum, psq, gamma[i], beta[i], sc, sc + C_);
    bn_lif_qkv<<<elemGrid, blk, 0, stream>>>(Y[i], sc, sc + C_, Sdst[i], i == 2 ? 1 : 0);
  }

  spike_attention<<<T_ * B_ * H_, blk, 0, stream>>>(Sq, Sk, SvT, O);

  lif_o<<<elemGrid, blk, 0, stream>>>(O, Sp);

  gemm16_f16split<2><<<gemmGrid, blk, 0, stream>>>(Sp, nullptr,
      whi + 3ull * nW, wlo + 3ull * nW, Yp);

  col_stats<<<dim3(8, 64), blk, 0, stream>>>(Yp, psum, psq);
  float* scp = ssbase + 3 * 2 * C_;
  bn_finalize<<<2, blk, 0, stream>>>(psum, psq, gamma[3], beta[3], scp, scp + C_);
  bn_lif_out<<<elemGrid, blk, 0, stream>>>(Yp, scp, scp + C_, (float*)d_out);
}
